// UDModel_72421738545878
// MI455X (gfx1250) — matrix operand order repaired and checked by reading, not compiled, not run on hardware
//
#include <hip/hip_runtime.h>

// ---------------------------------------------------------------------------
// CDNA5 (gfx1250) WMMA implementation of the UD biaffine parser head.
// wave32; v_wmma_f32_16x16x32_f16 for all GEMM-shaped work.
// ---------------------------------------------------------------------------

typedef _Float16 v4h  __attribute__((ext_vector_type(4)));
typedef _Float16 v8h  __attribute__((ext_vector_type(8)));
typedef _Float16 v16h __attribute__((ext_vector_type(16)));
typedef float    v8f  __attribute__((ext_vector_type(8)));

union F16Frag { v16h v; v8h h[2]; };

#define WMMA_F16(A, B, C) \
  __builtin_amdgcn_wmma_f32_16x16x32_f16(false, (A), false, (B), (short)0, (C), false, false)

#define NEG_INF (-__builtin_inff())

// Problem constants
#define BM 64
#define LL 128
#define DD 768
#define HH 1024
#define HA 256
#define PP 18
#define VV 40
#define MTOT (BM * LL) // 8192 tokens

// ---------------------------------------------------------------------------
// Generic tiled GEMM:  C[M,N] = act(A[M,K] @ B[K,N] + bias[N])
// AISHALF / NGUARD are compile-time so staging is straight-line vector code.
// Block tile 128x64, 8 waves (4m x 2n), each wave 32x32 = 2x2 WMMA tiles.
// ---------------------------------------------------------------------------
template <bool AISHALF, bool NGUARD>
__global__ __launch_bounds__(256) void gemm_wmma(
    const void* __restrict__ Aptr,
    const float* __restrict__ Bw, const float* __restrict__ bias,
    _Float16* __restrict__ out16, float* __restrict__ out32,
    int M, int N, int K, int doRelu)
{
  __shared__ __align__(16) _Float16 As[128 * 32];   // [row][k]
  __shared__ __align__(16) _Float16 Bt[64 * 32];    // [n][k] (transposed)

  const int tid  = threadIdx.x;
  const int lane = tid & 31;
  const int w    = tid >> 5;
  const int wm   = w & 3;         // wave m quadrant (32 rows)
  const int wn   = w >> 2;        // wave n half (32 cols)
  const int lm   = lane & 15;
  const int lh   = lane >> 4;
  const int m0   = blockIdx.y * 128;
  const int n0   = blockIdx.x * 64;

  const float*    Af = (const float*)Aptr;
  const _Float16* Ah = (const _Float16*)Aptr;

  v8f acc[2][2] = {};

  for (int kk = 0; kk < K; kk += 32) {
    // ---- Stage A tile 128x32 into LDS ----
    if (AISHALF) {
      // 512 x v8h copies (16B each), 2 per thread
      for (int i = 0; i < 2; ++i) {
        int idx = tid + 256 * i;              // 0..511
        int r = idx >> 2, kc = (idx & 3) * 8;
        *(v8h*)&As[r * 32 + kc] = *(const v8h*)(Ah + (long)(m0 + r) * K + kk + kc);
      }
    } else {
      // 1024 x float4 loads -> packed v4h stores, 4 per thread
      for (int i = 0; i < 4; ++i) {
        int idx = tid + 256 * i;              // 0..1023
        int r = idx >> 3, kc = (idx & 7) * 4;
        float4 f = *(const float4*)(Af + (long)(m0 + r) * K + kk + kc);
        v4h h = { (_Float16)f.x, (_Float16)f.y, (_Float16)f.z, (_Float16)f.w };
        *(v4h*)&As[r * 32 + kc] = h;
      }
    }
    // ---- Stage B tile transposed: Bt[n][k] ----
    // Each item: fixed column n, 4 consecutive k (coalesced across tid per k).
    for (int g = 0; g < 2; ++g) {
      int idx = tid + 256 * g;                // 0..511
      int nl = idx & 63, kg = idx >> 6;       // kg 0..7
      int n  = n0 + nl;
      float x0 = 0.f, x1 = 0.f, x2 = 0.f, x3 = 0.f;
      if (!NGUARD || n < N) {
        const float* bp = Bw + (long)(kk + kg * 4) * N + n;
        x0 = bp[0]; x1 = bp[N]; x2 = bp[2 * (long)N]; x3 = bp[3 * (long)N];
      }
      v4h h = { (_Float16)x0, (_Float16)x1, (_Float16)x2, (_Float16)x3 };
      *(v4h*)&Bt[nl * 32 + kg * 4] = h;
    }
    // Prefetch next K chunk into cache while this one computes
    if (kk + 32 < K) {
      if (AISHALF) __builtin_prefetch(Ah + (long)(m0 + (tid >> 1)) * K + kk + 32, 0, 1);
      else         __builtin_prefetch(Af + (long)(m0 + (tid >> 1)) * K + kk + 32, 0, 1);
      __builtin_prefetch(Bw + (long)(kk + 32 + (tid >> 6)) * N + n0 + (tid & 63), 0, 1);
    }
    __syncthreads();

    F16Frag a[2], bf[2];
    for (int mi = 0; mi < 2; ++mi) {
      const _Float16* p = &As[(wm * 32 + mi * 16 + lm) * 32 + lh * 8];
      a[mi].h[0] = *(const v8h*)p;            // K = koff .. koff+7
      a[mi].h[1] = *(const v8h*)(p + 16);     // K = koff+16 .. koff+23
    }
    for (int ni = 0; ni < 2; ++ni) {
      const _Float16* p = &Bt[(wn * 32 + ni * 16 + lm) * 32 + lh * 8];
      bf[ni].h[0] = *(const v8h*)p;           // K = koff .. koff+7
      bf[ni].h[1] = *(const v8h*)(p + 16);    // K = koff+16 .. koff+23
    }
    for (int mi = 0; mi < 2; ++mi)
      for (int ni = 0; ni < 2; ++ni)
        acc[mi][ni] = WMMA_F16(a[mi].v, bf[ni].v, acc[mi][ni]);
    __syncthreads();
  }

  // Epilogue: C lane mapping m = base + r + 8*lh, n = base + lm
  for (int mi = 0; mi < 2; ++mi)
    for (int ni = 0; ni < 2; ++ni)
      for (int r = 0; r < 8; ++r) {
        int m = m0 + wm * 32 + mi * 16 + r + 8 * lh;
        int n = n0 + wn * 32 + ni * 16 + lm;
        if (!NGUARD || n < N) {
          float val = acc[mi][ni][r] + bias[n];
          if (doRelu) val = val > 0.0f ? val : 0.0f;
          long o = (long)m * N + n;
          if (out16) out16[o] = (_Float16)val;
          if (out32) out32[o] = val;
        }
      }
}

// ---------------------------------------------------------------------------
// Per-batch arc scores: s[b,i,j] = dep[b,i,:] . Wh[b,j,:] + biases, masked.
// Contraction along last dim for both operands => fragments load straight
// from global f16 rows (no LDS). One block per batch; 8 waves (4m x 2n),
// each wave 32(m) x 64(n), K = 1024.
// ---------------------------------------------------------------------------
__global__ __launch_bounds__(256) void scores_wmma(
    const _Float16* __restrict__ dep16,  // [8192,1024]
    const _Float16* __restrict__ wh16,   // [8192,1024]
    const float* __restrict__ headBias,  // [8192]
    const float* __restrict__ depBias,   // [8192]
    const float* __restrict__ bias1,     // [1]
    const float* __restrict__ padMask,   // [64,128]
    float* __restrict__ out)             // [64,128,128]
{
  const int b    = blockIdx.x;
  const int tid  = threadIdx.x;
  const int lane = tid & 31;
  const int w    = tid >> 5;
  const int wm   = w & 3;
  const int wn   = w >> 2;
  const int lm   = lane & 15;
  const int lh   = lane >> 4;

  v8f acc[2][4] = {};

  for (int kk = 0; kk < HH; kk += 32) {
    F16Frag a[2], bf[4];
    for (int mi = 0; mi < 2; ++mi) {
      long t = (long)b * LL + wm * 32 + mi * 16 + lm;
      const _Float16* p = dep16 + t * HH + kk + lh * 8;
      a[mi].h[0] = *(const v8h*)p;
      a[mi].h[1] = *(const v8h*)(p + 16);
    }
    for (int ni = 0; ni < 4; ++ni) {
      long j = (long)b * LL + wn * 64 + ni * 16 + lm;
      const _Float16* p = wh16 + j * HH + kk + lh * 8;
      bf[ni].h[0] = *(const v8h*)p;
      bf[ni].h[1] = *(const v8h*)(p + 16);
    }
    for (int mi = 0; mi < 2; ++mi)
      for (int ni = 0; ni < 4; ++ni)
        acc[mi][ni] = WMMA_F16(a[mi].v, bf[ni].v, acc[mi][ni]);
  }

  const float bb = bias1[0];
  for (int mi = 0; mi < 2; ++mi)
    for (int ni = 0; ni < 4; ++ni)
      for (int r = 0; r < 8; ++r) {
        int i = wm * 32 + mi * 16 + r + 8 * lh;
        int j = wn * 64 + ni * 16 + lm;
        float val = acc[mi][ni][r] + headBias[b * LL + j] + depBias[b * LL + i] + bb;
        if (!(padMask[b * LL + j] > 0.0f)) val = NEG_INF;
        out[((long)b * LL + i) * LL + j] = val;
      }
}

// ---------------------------------------------------------------------------
// Labeled bilinear, one label v per blockIdx.y:
//   s[t,v] = sum_e (G[t,:] @ W_v[:, e]) * Dep[t,e]  (+ bias terms)
// Block handles 128 tokens over full 256x256 W_v. 8 waves, each owns 16 rows;
// acc over 16 n-tiles; fused multiply-by-Dep row-reduction epilogue.
// ---------------------------------------------------------------------------
__global__ __launch_bounds__(256) void bilinear_wmma(
    const _Float16* __restrict__ gat16,   // [8192,256]  gathered head reps
    const _Float16* __restrict__ dep16,   // [8192,256]  arc dep reps
    const float* __restrict__ Warc,       // [40,256,256]
    const float* __restrict__ harcB,      // [8192,40]
    const float* __restrict__ darcB,      // [8192,40]
    const float* __restrict__ biasArc,    // [40]
    float* __restrict__ out)              // [8192,40]
{
  __shared__ __align__(16) _Float16 Bt[256 * 32];  // [e][k] transposed W_v chunk
  __shared__ float ssum[128];

  const int v    = blockIdx.y;
  const int m0   = blockIdx.x * 128;
  const int tid  = threadIdx.x;
  const int lane = tid & 31;
  const int w    = tid >> 5;
  const int lm   = lane & 15;
  const int lh   = lane >> 4;

  if (tid < 128) ssum[tid] = 0.0f;

  v8f acc[16] = {};
  const long wbase = (long)v * HA * HA;

  for (int kk = 0; kk < HA; kk += 32) {
    // Stage Bt[e][k] = W_v[k][e]: fixed column e per item, 4 consecutive k,
    // 4 coalesced b32 loads + one packed ds_store_b64.
    for (int i = 0; i < 8; ++i) {
      int idx = tid + 256 * i;               // 0..2047
      int e = idx & 255, kg = idx >> 8;      // kg 0..7
      const float* bp = Warc + wbase + (long)(kk + kg * 4) * HA + e;
      v4h h = { (_Float16)bp[0], (_Float16)bp[HA], (_Float16)bp[2 * HA], (_Float16)bp[3 * HA] };
      *(v4h*)&Bt[e * 32 + kg * 4] = h;
    }
    if (kk + 32 < HA)
      __builtin_prefetch(Warc + wbase + (long)(kk + 32) * HA + tid, 0, 1);
    __syncthreads();

    F16Frag a;
    {
      long t = m0 + w * 16 + lm;
      const _Float16* p = gat16 + t * HA + kk + lh * 8;
      a.h[0] = *(const v8h*)p;
      a.h[1] = *(const v8h*)(p + 16);
    }
    for (int nt = 0; nt < 16; ++nt) {
      F16Frag bf;
      const _Float16* p = &Bt[(nt * 16 + lm) * 32 + lh * 8];
      bf.h[0] = *(const v8h*)p;
      bf.h[1] = *(const v8h*)(p + 16);
      acc[nt] = WMMA_F16(a.v, bf.v, acc[nt]);
    }
    __syncthreads();
  }

  // epilogue: s[t] += C[t, e] * Dep[t, e], reduce over e via LDS
  for (int nt = 0; nt < 16; ++nt)
    for (int r = 0; r < 8; ++r) {
      int ml = r + 8 * lh;
      int n  = nt * 16 + lm;
      long t = m0 + w * 16 + ml;
      float d = (float)dep16[t * HA + n];
      atomicAdd(&ssum[w * 16 + ml], acc[nt][r] * d);
    }
  __syncthreads();

  if (tid < 128) {
    long t = m0 + tid;
    out[t * VV + v] = ssum[tid] + harcB[t * VV + v] + darcB[t * VV + v] + biasArc[v];
  }
}

// ---------------------------------------------------------------------------
// Small helper kernels (tail work, <1% of FLOPs)
// ---------------------------------------------------------------------------
__global__ void gather_k(const _Float16* __restrict__ src,  // [8192,256]
                         const int* __restrict__ heads,     // [8192]
                         _Float16* __restrict__ dst)        // [8192,256]
{
  long idx = (long)blockIdx.x * blockDim.x + threadIdx.x;   // over 8192*32 items
  int t = (int)(idx >> 5), c = (int)(idx & 31) * 8;
  int b = t >> 7;
  int h = heads[t];
  v8h val = {};
  if (h >= 0) {
    int hh = h > (LL - 1) ? (LL - 1) : h;
    val = *(const v8h*)(src + ((long)(b * LL + hh)) * HA + c);
  }
  *(v8h*)(dst + (long)t * HA + c) = val;
}

__global__ void rowdot_k(const _Float16* __restrict__ Mtx, const float* __restrict__ vec,
                         float* __restrict__ out, int K)
{
  int t = blockIdx.x * blockDim.x + threadIdx.x;
  const _Float16* row = Mtx + (long)t * K;
  float s = 0.0f;
  for (int k = 0; k < K; ++k) s += (float)row[k] * vec[k];
  out[t] = s;
}

__global__ void arcbias_k(const _Float16* __restrict__ Mtx, // [8192,256]
                          const float* __restrict__ vecVA,  // [256,40]
                          float* __restrict__ out)          // [8192,40]
{
  long g = (long)blockIdx.x * blockDim.x + threadIdx.x;     // over 8192*40
  int t = (int)(g / VV), v = (int)(g % VV);
  const _Float16* row = Mtx + (long)t * HA;
  float s = 0.0f;
  for (int d = 0; d < HA; ++d) s += (float)row[d] * vecVA[d * VV + v];
  out[g] = s;
}

// ---------------------------------------------------------------------------
extern "C" void kernel_launch(void* const* d_in, const int* in_sizes, int n_in,
                              void* d_out, int out_size, void* d_ws, size_t ws_size,
                              hipStream_t stream)
{
  (void)in_sizes; (void)n_in; (void)out_size; (void)ws_size;

  const float* ann      = (const float*)d_in[0];
  const float* padMask  = (const float*)d_in[1];
  const int*   heads    = (const int*)d_in[2];
  const float* W_head   = (const float*)d_in[3];
  const float* b_head   = (const float*)d_in[4];
  const float* W_dep    = (const float*)d_in[5];
  const float* b_dep    = (const float*)d_in[6];
  const float* W_lin    = (const float*)d_in[7];
  const float* b_lin    = (const float*)d_in[8];
  const float* head_vec = (const float*)d_in[9];
  const float* dep_vec  = (const float*)d_in[10];
  const float* bias1    = (const float*)d_in[11];
  const float* W_headA  = (const float*)d_in[12];
  const float* b_headA  = (const float*)d_in[13];
  const float* W_depA   = (const float*)d_in[14];
  const float* b_depA   = (const float*)d_in[15];
  const float* W_arc    = (const float*)d_in[16];
  const float* hvA      = (const float*)d_in[17];
  const float* dvA      = (const float*)d_in[18];
  const float* biasA    = (const float*)d_in[19];
  const float* W_pos    = (const float*)d_in[20];
  const float* b_pos    = (const float*)d_in[21];

  // Workspace layout (f16 intermediates for WMMA consumers), ~64.1 MB total
  char* ws = (char*)d_ws;
  _Float16* annH16  = (_Float16*)(ws + 0);                       // 16 MB
  _Float16* annD16  = (_Float16*)(ws + (16l << 20));             // 16 MB
  _Float16* Wh16    = (_Float16*)(ws + (32l << 20));             // 16 MB
  _Float16* annHA16 = (_Float16*)(ws + (48l << 20));             //  4 MB
  _Float16* depA16  = (_Float16*)(ws + (52l << 20));             //  4 MB
  _Float16* gat16   = (_Float16*)(ws + (56l << 20));             //  4 MB
  float* headBias   = (float*)(ws + (60l << 20));                // 32 KB
  float* depBias    = (float*)(ws + (60l << 20) + (64l << 10));  // 32 KB
  float* harcB      = (float*)(ws + (60l << 20) + (128l << 10));              // 1.25 MB
  float* darcB      = (float*)(ws + (60l << 20) + (128l << 10) + (2l << 20)); // 1.25 MB

  float* pos_out    = (float*)d_out;                         // [64,128,18]
  float* scores_out = pos_out + (long)BM * LL * PP;          // [64,128,128]
  float* arc_out    = scores_out + (long)BM * LL * LL;       // [64,128,40]

  dim3 blk(256);
  const int M = MTOT;

  // MLP / linear GEMMs (unguarded: N multiple of 64)
  gemm_wmma<false, false><<<dim3(HH / 64, M / 128), blk, 0, stream>>>(ann, W_head, b_head, annH16, nullptr, M, HH, DD, 1);
  gemm_wmma<false, false><<<dim3(HH / 64, M / 128), blk, 0, stream>>>(ann, W_dep,  b_dep,  annD16, nullptr, M, HH, DD, 1);
  gemm_wmma<true,  false><<<dim3(HH / 64, M / 128), blk, 0, stream>>>(annH16, W_lin, b_lin, Wh16, nullptr, M, HH, HH, 0);
  gemm_wmma<false, false><<<dim3(HA / 64, M / 128), blk, 0, stream>>>(ann, W_headA, b_headA, annHA16, nullptr, M, HA, DD, 1);
  gemm_wmma<false, false><<<dim3(HA / 64, M / 128), blk, 0, stream>>>(ann, W_depA,  b_depA,  depA16, nullptr, M, HA, DD, 1);
  // POS head: N=18 -> guarded instantiation
  gemm_wmma<false, true><<<dim3(1, M / 128), blk, 0, stream>>>(ann, W_pos, b_pos, nullptr, pos_out, M, PP, DD, 0);

  // bias dots, gather, per-label bias GEMVs
  rowdot_k<<<M / 256, blk, 0, stream>>>(annH16, head_vec, headBias, HH);
  rowdot_k<<<M / 256, blk, 0, stream>>>(annD16, dep_vec,  depBias,  HH);
  gather_k<<<(M * 32) / 256, blk, 0, stream>>>(annHA16, heads, gat16);
  arcbias_k<<<(M * VV) / 256, blk, 0, stream>>>(gat16,  hvA, harcB);
  arcbias_k<<<(M * VV) / 256, blk, 0, stream>>>(depA16, dvA, darcB);

  // biaffine score matrices and labeled bilinear
  scores_wmma<<<BM, blk, 0, stream>>>(annD16, Wh16, headBias, depBias, bias1, padMask, scores_out);
  bilinear_wmma<<<dim3(M / 128, VV), blk, 0, stream>>>(gat16, depA16, W_arc, harcB, darcB, biasA, arc_out);
}
